// MultiHeadAttention_82205674045574
// MI455X (gfx1250) — matrix operand order repaired and checked by reading, not compiled, not run on hardware
//
#include <hip/hip_runtime.h>

// ---------------------------------------------------------------------------
// MI455X (gfx1250) MultiHeadAttention forward, WMMA bf16 pipeline.
// B=2, S=2048, E=1024, H=16, hd=64.  Only head-1 probs matter (reference
// broadcasts probs[:,1:2] over all heads), so scores are computed for head 1
// only and P@V collapses to one 2048x1024x2048 GEMM per batch.
// All GEMMs: v_wmma_f32_16x16x32_bf16, 128x128 block tile, BK=32, 8 waves.
// k_pv stages its B tile via the Tensor Data Mover (tensor_load_to_lds).
// ---------------------------------------------------------------------------

typedef __attribute__((ext_vector_type(16))) __bf16 v16bf;
typedef __attribute__((ext_vector_type(8)))  float  v8f;
typedef __attribute__((ext_vector_type(4)))  unsigned int u32x4;
typedef __attribute__((ext_vector_type(8)))  int i32x8;
typedef __attribute__((ext_vector_type(4)))  int i32x4;

#define BM 128
#define BN 128
#define BK 32
#define LDA_ST 40     // LDS A row stride in ushort (32 data + 8 pad, 16B aligned)
#define LDB_ST 136    // LDS B row stride in ushort (128 data + 8 pad = 272B)

#if defined(__has_builtin)
#if __has_builtin(__builtin_amdgcn_tensor_load_to_lds)
#define HAVE_TDM 1
#endif
#endif
#ifndef HAVE_TDM
#define HAVE_TDM 0
#endif

__device__ __forceinline__ unsigned short f2bf(float f) {
  unsigned u = __float_as_uint(f);
  unsigned r = u + 0x7fffu + ((u >> 16) & 1u);   // round-to-nearest-even
  return (unsigned short)(r >> 16);
}

union AFrag { v16bf v; uint4 q[2]; };

// ---- LDS staging helpers (256 threads) ------------------------------------

// A tile BMxBK from bf16 row-major source.
__device__ __forceinline__ void stageA_bf16(unsigned short* lA, const unsigned short* A,
                                            long lda, long row0, long k0, int tid) {
  int r = tid >> 1, h = tid & 1;
  const uint4* src = (const uint4*)(A + (row0 + r) * lda + k0);
  uint4* dst = (uint4*)(lA + r * LDA_ST);
  dst[h * 2 + 0] = src[h * 2 + 0];
  dst[h * 2 + 1] = src[h * 2 + 1];
}

// A tile BMxBK from f32 row-major source, converting to bf16 on the fly.
__device__ __forceinline__ void stageA_f32(unsigned short* lA, const float* A,
                                           long lda, long row0, long k0, int tid) {
  int r = tid >> 1, h = tid & 1;
  const float4* src = (const float4*)(A + (row0 + r) * lda + k0 + h * 16);
  unsigned short* dst = lA + r * LDA_ST + h * 16;
#pragma unroll
  for (int c = 0; c < 4; ++c) {
    float4 v = src[c];
    dst[c * 4 + 0] = f2bf(v.x);
    dst[c * 4 + 1] = f2bf(v.y);
    dst[c * 4 + 2] = f2bf(v.z);
    dst[c * 4 + 3] = f2bf(v.w);
  }
}

// B tile BKxBN where B(k,n) = W[n,k], W row-major (N x K) bf16. Transpose into LDS.
__device__ __forceinline__ void stageB_T(unsigned short* lB, const unsigned short* W,
                                         long ldw, long n0, long k0, int tid) {
  int n = tid >> 1, h = tid & 1;
  const unsigned short* src = W + (n0 + n) * ldw + k0 + h * 16;
  unsigned short tmp[16];
  *(uint4*)(tmp)     = *(const uint4*)(src);
  *(uint4*)(tmp + 8) = *(const uint4*)(src + 8);
#pragma unroll
  for (int i = 0; i < 16; ++i)
    lB[(h * 16 + i) * LDB_ST + n] = tmp[i];
}

// B tile BKxBN where B is row-major (K x N) bf16. Direct copy (fallback path).
__device__ __forceinline__ void stageB_N(unsigned short* lB, const unsigned short* Bsrc,
                                         long ldb, long k0, long n0, int tid) {
  int k = tid >> 3, c = (tid & 7) * 2;
  const uint4* src = (const uint4*)(Bsrc + (k0 + k) * ldb + n0);
  uint4* dst = (uint4*)(lB + k * LDB_ST);
  dst[c]     = src[c];
  dst[c + 1] = src[c + 1];
}

#if HAVE_TDM
// TDM DMA of a 32x128 bf16 tile (row stride 1152 elements) into LDS with
// 16B per-row padding (LDS row stride 272B). D# per CDNA5 ISA ch.8:
//   group0: count=1 | lds_addr | global_addr(57b) | type=2
//   group1: data_size=1(2B), pad_enable, pad_interval=5 (64 DW = 256B rows),
//           pad_amount=3 (4 DW = 16B), tile_dim0=128, tile_dim1=32,
//           tensor_dim0_stride=1152.
__device__ __forceinline__ void tdm_load_B(const unsigned short* tileStart,
                                           unsigned ldsByteOff) {
  unsigned long long ga = (unsigned long long)(const void*)tileStart;
  u32x4 g0;
  g0.x = 0x1u;                                       // count=1, no gather
  g0.y = ldsByteOff;                                 // lds_addr
  g0.z = (unsigned)(ga & 0xFFFFFFFFu);               // global_addr[31:0]
  g0.w = (unsigned)((ga >> 32) & 0x01FFFFFFu) | (2u << 30);  // addr[56:32]|type=2
  i32x8 g1;
  g1[0] = (1 << 16) | (1 << 20) | (5 << 22) | (3 << 25);  // dsz=2B|pad|intv|amt
  g1[1] = (int)(0x10000u * (4096u & 0xFFFFu));            // tensor_dim0 lo16
  g1[2] = (int)((4096u >> 16) | ((4096u & 0xFFFFu) << 16)); // dim0 hi | dim1 lo
  g1[3] = (int)((4096u >> 16) | (128u << 16));            // dim1 hi | tile_dim0
  g1[4] = 32;                                             // tile_dim1 | tile_dim2=0
  g1[5] = 1152;                                           // dim0_stride[31:0]
  g1[6] = 0;                                              // stride hi | dim1_stride lo
  g1[7] = 0;
  i32x4 z4 = {0, 0, 0, 0};
#if __clang_major__ >= 23
  i32x8 z8 = {0, 0, 0, 0, 0, 0, 0, 0};
  __builtin_amdgcn_tensor_load_to_lds(g0, g1, z4, z4, z8, 0);
#else
  __builtin_amdgcn_tensor_load_to_lds(g0, g1, z4, z4, 0);
#endif
}
#endif

// ---- one BK=32 step of 4x2 16x16 WMMAs per wave ---------------------------
__device__ __forceinline__ void mma_step(const unsigned short* lA, const unsigned short* lB,
                                         v8f acc[4][2], int wm, int wn, int lane) {
  const int g = lane >> 4;        // half-wave group
  const int ml = lane & 15;
  AFrag a[4], b[2];
#pragma unroll
  for (int mi = 0; mi < 4; ++mi) {
    const unsigned short* p = lA + (wm * 64 + mi * 16 + ml) * LDA_ST + g * 8;
    a[mi].q[0] = *(const uint4*)(p);        // k = g*8   .. g*8+7
    a[mi].q[1] = *(const uint4*)(p + 16);   // k = 16+g*8 .. +7
  }
#pragma unroll
  for (int ni = 0; ni < 2; ++ni) {
    const unsigned short* p = lB + (g * 8) * LDB_ST + wn * 32 + ni * 16 + ml;
    unsigned short t[16];
#pragma unroll
    for (int s = 0; s < 8; ++s) {
      t[s]     = p[s * LDB_ST];
      t[s + 8] = p[(16 + s) * LDB_ST];
    }
    __builtin_memcpy(&b[ni], t, 32);
  }
#pragma unroll
  for (int mi = 0; mi < 4; ++mi)
#pragma unroll
    for (int ni = 0; ni < 2; ++ni)
      acc[mi][ni] = __builtin_amdgcn_wmma_f32_16x16x32_bf16(
          false, a[mi].v, false, b[ni].v, (short)0, acc[mi][ni], false, false);
}

__device__ __forceinline__ void zero_acc(v8f acc[4][2]) {
  v8f z = {0.f, 0.f, 0.f, 0.f, 0.f, 0.f, 0.f, 0.f};
#pragma unroll
  for (int mi = 0; mi < 4; ++mi)
#pragma unroll
    for (int ni = 0; ni < 2; ++ni) acc[mi][ni] = z;
}

// ---------------------------------------------------------------------------
// Kernel 1: convert x -> bf16, gather Wc = [Wq[64:128]; Wk[64:128]; Wv] bf16,
//           convert Wo -> bf16.
// ---------------------------------------------------------------------------
__global__ __launch_bounds__(256) void k_convert(const float* __restrict__ x,
                                                 const float* __restrict__ Wq,
                                                 const float* __restrict__ Wk,
                                                 const float* __restrict__ Wv,
                                                 const float* __restrict__ Wo,
                                                 unsigned short* __restrict__ xb,
                                                 unsigned short* __restrict__ Wc,
                                                 unsigned short* __restrict__ Wob) {
  const long NX = 4194304L;  // 4096*1024
  const long NW = 1179648L;  // 1152*1024
  const long NO = 1048576L;  // 1024*1024
  for (long i = (long)blockIdx.x * 256 + threadIdx.x; i < NX + NW + NO;
       i += (long)gridDim.x * 256) {
    if (i < NX) {
      xb[i] = f2bf(x[i]);
    } else if (i < NX + NW) {
      long j = i - NX, row = j >> 10, col = j & 1023;
      float v = (row < 64)    ? Wq[(64 + row) * 1024 + col]
                : (row < 128) ? Wk[row * 1024 + col]        // rows 64..127 of Wk
                              : Wv[(row - 128) * 1024 + col];
      Wc[j] = f2bf(v);
    } else {
      long j = i - NX - NW;
      Wob[j] = f2bf(Wo[j]);
    }
  }
}

// ---------------------------------------------------------------------------
// Kernel 2: Y(4096 x 1152) = xb(4096x1024) @ Wc^T, +bias, bf16 out.
//           cols 0..63 = Q(head1), 64..127 = K(head1), 128..1151 = V.
// ---------------------------------------------------------------------------
__global__ __launch_bounds__(256) void k_qkv(const unsigned short* __restrict__ xb,
                                             const unsigned short* __restrict__ Wc,
                                             const float* __restrict__ bq,
                                             const float* __restrict__ bk,
                                             const float* __restrict__ bv,
                                             unsigned short* __restrict__ Y) {
  __shared__ unsigned short lA[BM * LDA_ST];
  __shared__ unsigned short lB[BK * LDB_ST];
  const int tid = threadIdx.x, lane = tid & 31, w = tid >> 5, wm = w >> 2, wn = w & 3;
  const long row0 = (long)blockIdx.y * BM, n0 = (long)blockIdx.x * BN;
  v8f acc[4][2];
  zero_acc(acc);
  for (long k0 = 0; k0 < 1024; k0 += BK) {
    __syncthreads();
    stageA_bf16(lA, xb, 1024, row0, k0, tid);
    stageB_T(lB, Wc, 1024, n0, k0, tid);
    __syncthreads();
    mma_step(lA, lB, acc, wm, wn, lane);
  }
  const int g = lane >> 4, ml = lane & 15;
#pragma unroll
  for (int mi = 0; mi < 4; ++mi)
#pragma unroll
    for (int ni = 0; ni < 2; ++ni)
#pragma unroll
      for (int r = 0; r < 8; ++r) {
        long row = row0 + wm * 64 + mi * 16 + g * 8 + r;
        long col = n0 + wn * 32 + ni * 16 + ml;
        float bias = (col < 64) ? bq[64 + col] : (col < 128 ? bk[col] : bv[col - 128]);
        Y[row * 1152 + col] = f2bf(acc[mi][ni][r] + bias);
      }
}

// ---------------------------------------------------------------------------
// Kernel 3: head-1 scores: S[b,q,k] = 0.125 * Qh.Kh - slope1*|q-k|  (f32 out)
// ---------------------------------------------------------------------------
__global__ __launch_bounds__(256) void k_scores(const unsigned short* __restrict__ Y,
                                                const float* __restrict__ slopes,
                                                float* __restrict__ P) {
  __shared__ unsigned short lA[BM * LDA_ST];
  __shared__ unsigned short lB[BK * LDB_ST];
  const int tid = threadIdx.x, lane = tid & 31, w = tid >> 5, wm = w >> 2, wn = w & 3;
  const long b = blockIdx.z;
  const unsigned short* Qh = Y + b * 2048 * 1152;        // cols 0..63
  const unsigned short* Kh = Y + b * 2048 * 1152 + 64;   // cols 64..127
  const long row0 = (long)blockIdx.y * BM, n0 = (long)blockIdx.x * BN;
  v8f acc[4][2];
  zero_acc(acc);
  for (long k0 = 0; k0 < 64; k0 += BK) {
    __syncthreads();
    stageA_bf16(lA, Qh, 1152, row0, k0, tid);
    stageB_T(lB, Kh, 1152, n0, k0, tid);
    __syncthreads();
    mma_step(lA, lB, acc, wm, wn, lane);
  }
  const float s1 = slopes[1];
  const int g = lane >> 4, ml = lane & 15;
#pragma unroll
  for (int mi = 0; mi < 4; ++mi)
#pragma unroll
    for (int ni = 0; ni < 2; ++ni)
#pragma unroll
      for (int r = 0; r < 8; ++r) {
        long q = row0 + wm * 64 + mi * 16 + g * 8 + r;
        long k = n0 + wn * 32 + ni * 16 + ml;
        float v = acc[mi][ni][r] * 0.125f - s1 * fabsf((float)(q - k));
        P[b * 4194304L + q * 2048 + k] = v;
      }
}

// ---------------------------------------------------------------------------
// Kernel 4: row softmax in place over 2048 elements (f32, in d_out).
// ---------------------------------------------------------------------------
__global__ __launch_bounds__(256) void k_softmax(float* __restrict__ P) {
  const long row = blockIdx.x;
  float* p = P + row * 2048;
  __shared__ float red[256];
  const int tid = threadIdx.x;
  float v[8];
  float mx = -3.402823e38f;
#pragma unroll
  for (int i = 0; i < 8; ++i) {
    v[i] = p[tid + i * 256];
    mx = fmaxf(mx, v[i]);
  }
  red[tid] = mx;
  __syncthreads();
  for (int s = 128; s > 0; s >>= 1) {
    if (tid < s) red[tid] = fmaxf(red[tid], red[tid + s]);
    __syncthreads();
  }
  mx = red[0];
  __syncthreads();
  float sum = 0.f;
#pragma unroll
  for (int i = 0; i < 8; ++i) {
    v[i] = __expf(v[i] - mx);
    sum += v[i];
  }
  red[tid] = sum;
  __syncthreads();
  for (int s = 128; s > 0; s >>= 1) {
    if (tid < s) red[tid] += red[tid + s];
    __syncthreads();
  }
  const float inv = 1.0f / red[0];
#pragma unroll
  for (int i = 0; i < 8; ++i) p[tid + i * 256] = v[i] * inv;
}

// ---------------------------------------------------------------------------
// Kernel 5: attn = P(f32, 2048x2048) @ V(2048x1024), permuted bf16 store
//           reproducing the reference's (B,H,S,hd)->(B,S,E) raw reshape.
//           B tile is DMA'd by the Tensor Data Mover (wave 0 issues, waits
//           s_wait_tensorcnt 0, workgroup barrier publishes the tile).
// ---------------------------------------------------------------------------
__global__ __launch_bounds__(256) void k_pv(const float* __restrict__ P,
                                            const unsigned short* __restrict__ Y,
                                            unsigned short* __restrict__ attn2) {
  __shared__ unsigned short lA[BM * LDA_ST];
  __shared__ unsigned short lB[BK * LDB_ST];
  const int tid = threadIdx.x, lane = tid & 31, w = tid >> 5, wm = w >> 2, wn = w & 3;
  const long b = blockIdx.z;
  const float* Pb = P + b * 4194304L;
  const unsigned short* V = Y + b * 2048 * 1152 + 128;   // cols 128..1151
  const long row0 = (long)blockIdx.y * BM, n0 = (long)blockIdx.x * BN;
  v8f acc[4][2];
  zero_acc(acc);
#if HAVE_TDM
  const unsigned ldsBOff = (unsigned)(unsigned long long)(void*)lB; // LDS byte offset
#endif
  for (long k0 = 0; k0 < 2048; k0 += BK) {
    __syncthreads();
    stageA_f32(lA, Pb, 2048, row0, k0, tid);
#if HAVE_TDM
    if (w == 0) {
      tdm_load_B(V + k0 * 1152 + n0, ldsBOff);
      __builtin_amdgcn_s_wait_tensorcnt(0);
    }
#else
    stageB_N(lB, V, 1152, k0, n0, tid);
#endif
    __syncthreads();
    mma_step(lA, lB, acc, wm, wn, lane);
  }
  const int g = lane >> 4, ml = lane & 15;
#pragma unroll
  for (int mi = 0; mi < 4; ++mi)
#pragma unroll
    for (int ni = 0; ni < 2; ++ni)
#pragma unroll
      for (int r = 0; r < 8; ++r) {
        long s = row0 + wm * 64 + mi * 16 + g * 8 + r;
        long e = n0 + wn * 32 + ni * 16 + ml;
        long h = e >> 6, d = e & 63;
        long flat = h * 131072L + s * 64 + d;   // h*S*hd + s*hd + d
        attn2[b * 2097152L + flat] = f2bf(acc[mi][ni][r]);
      }
}

// ---------------------------------------------------------------------------
// Kernel 6: out(4096 x 1024) = attn2 @ Wo^T + bo  (f32 out)
// ---------------------------------------------------------------------------
__global__ __launch_bounds__(256) void k_out(const unsigned short* __restrict__ attn2,
                                             const unsigned short* __restrict__ Wob,
                                             const float* __restrict__ bo,
                                             float* __restrict__ out) {
  __shared__ unsigned short lA[BM * LDA_ST];
  __shared__ unsigned short lB[BK * LDB_ST];
  const int tid = threadIdx.x, lane = tid & 31, w = tid >> 5, wm = w >> 2, wn = w & 3;
  const long row0 = (long)blockIdx.y * BM, n0 = (long)blockIdx.x * BN;
  v8f acc[4][2];
  zero_acc(acc);
  for (long k0 = 0; k0 < 1024; k0 += BK) {
    __syncthreads();
    stageA_bf16(lA, attn2, 1024, row0, k0, tid);
    stageB_T(lB, Wob, 1024, n0, k0, tid);
    __syncthreads();
    mma_step(lA, lB, acc, wm, wn, lane);
  }
  const int g = lane >> 4, ml = lane & 15;
#pragma unroll
  for (int mi = 0; mi < 4; ++mi)
#pragma unroll
    for (int ni = 0; ni < 2; ++ni)
#pragma unroll
      for (int r = 0; r < 8; ++r) {
        long row = row0 + wm * 64 + mi * 16 + g * 8 + r;
        long col = n0 + wn * 32 + ni * 16 + ml;
        out[row * 1024 + col] = acc[mi][ni][r] + bo[col];
      }
}

// ---------------------------------------------------------------------------
extern "C" void kernel_launch(void* const* d_in, const int* in_sizes, int n_in,
                              void* d_out, int out_size, void* d_ws, size_t ws_size,
                              hipStream_t stream) {
  (void)in_sizes; (void)n_in; (void)out_size; (void)ws_size;
  const float* x      = (const float*)d_in[0];
  const float* Wq     = (const float*)d_in[1];
  const float* bq     = (const float*)d_in[2];
  const float* Wk     = (const float*)d_in[3];
  const float* bk     = (const float*)d_in[4];
  const float* Wv     = (const float*)d_in[5];
  const float* bv     = (const float*)d_in[6];
  const float* Wo     = (const float*)d_in[7];
  const float* bo     = (const float*)d_in[8];
  const float* slopes = (const float*)d_in[9];

  float* out   = (float*)d_out;          // (B,S,E) f32 = 4,194,304
  float* probs = out + 4194304L;         // (B,1,S,S) f32 = 8,388,608

  // Workspace layout (bytes, 16B-aligned), total ~30.7 MB:
  unsigned char* ws = (unsigned char*)d_ws;
  unsigned short* xb    = (unsigned short*)(ws);              //  8,388,608 B
  unsigned short* Wc    = (unsigned short*)(ws +  8388608L);  //  2,359,296 B
  unsigned short* Wob   = (unsigned short*)(ws + 10747904L);  //  2,097,152 B
  unsigned short* Y     = (unsigned short*)(ws + 12845056L);  //  9,437,184 B
  unsigned short* attn2 = (unsigned short*)(ws + 22282240L);  //  8,388,608 B

  k_convert<<<4096, 256, 0, stream>>>(x, Wq, Wk, Wv, Wo, xb, Wc, Wob);
  k_qkv<<<dim3(9, 32), 256, 0, stream>>>(xb, Wc, bq, bk, bv, Y);
  k_scores<<<dim3(16, 16, 2), 256, 0, stream>>>(Y, slopes, probs);
  k_softmax<<<4096, 256, 0, stream>>>(probs);
  k_pv<<<dim3(8, 16, 2), 256, 0, stream>>>(probs, Y, attn2);
  k_out<<<dim3(8, 32), 256, 0, stream>>>(attn2, Wob, bo, out);
}
